// CodeUpdater_22058952032956
// MI455X (gfx1250) — matrix operand order repaired and checked by reading, not compiled, not run on hardware
//
#include <hip/hip_runtime.h>
#include <hip/hip_bf16.h>

// Problem constants (match reference)
#define NN 4096
#define MM 8192
#define KK 32768
#define RR 8
#define DD 512
#define HH 256
#define TWO_D 1024   // 2*D
#define FOUR_H 1024  // 4*H

typedef __attribute__((ext_vector_type(16))) __bf16 v16bf;
typedef __attribute__((ext_vector_type(8)))  float  v8f;
typedef __attribute__((ext_vector_type(2)))  __bf16 bf16x2;

// Fast activations: v_rcp_f32 instead of the IEEE divide chain; tanh via
// exp2-based __expf + rcp (clamped so e^{2x} cannot overflow).
__device__ __forceinline__ float fast_rcp(float x) { return __builtin_amdgcn_rcpf(x); }
__device__ __forceinline__ float sigmoidf_(float x) {
    return fast_rcp(1.0f + __expf(-x));
}
__device__ __forceinline__ float tanhf_(float x) {
    float cx = fminf(fmaxf(x, -15.0f), 15.0f);
    float e  = __expf(2.0f * cx);
    return (e - 1.0f) * fast_rcp(e + 1.0f);
}

// ---------------------------------------------------------------------------
// small prep kernels
// ---------------------------------------------------------------------------
__global__ __launch_bounds__(256) void zero_f32(float* __restrict__ p, int n) {
    int i = blockIdx.x * blockDim.x + threadIdx.x;
    if (i < n) p[i] = 0.0f;
}

__global__ __launch_bounds__(256) void f32_to_bf16(const float* __restrict__ in,
                                                   __bf16* __restrict__ out, int n) {
    int i = blockIdx.x * blockDim.x + threadIdx.x;
    if (i < n) out[i] = (__bf16)in[i];
}

// Whh [4H=1024][H=256] f32  ->  packed bf16 pairs, transposed:
// w2[jj*1024 + i] = pack( Whh[i][2*jj], Whh[i][2*jj+1] ),  jj in [0,128), i in [0,1024)
__global__ __launch_bounds__(256) void pack_whh(const float* __restrict__ whh,
                                                unsigned int* __restrict__ w2) {
    int idx = blockIdx.x * blockDim.x + threadIdx.x;
    if (idx < 128 * FOUR_H) {
        int jj = idx >> 10;
        int i  = idx & 1023;
        bf16x2 p;
        p.x = (__bf16)whh[i * HH + 2 * jj];
        p.y = (__bf16)whh[i * HH + 2 * jj + 1];
        w2[idx] = __builtin_bit_cast(unsigned int, p);
    }
}

// x_bf16[n][0:512] = code[n], x_bf16[n][512:1024] = ctu[n]
__global__ __launch_bounds__(256) void build_x(const float* __restrict__ code,
                                               const float* __restrict__ ctu,
                                               __bf16* __restrict__ xbf) {
    int idx = blockIdx.x * blockDim.x + threadIdx.x;
    if (idx < NN * TWO_D) {
        int n = idx >> 10;
        int j = idx & 1023;
        float v = (j < DD) ? code[n * DD + j] : ctu[n * DD + (j - DD)];
        xbf[idx] = (__bf16)v;
    }
}

// ---------------------------------------------------------------------------
// Tiling: block = 256 threads (8 waves), block tile 64(M) x 128(N), BK = 32.
// Waves in 2(M) x 4(N); each wave owns a 32x32 output tile:
//   2 A-fragments + 2 B-fragments -> 4 v_wmma per k-step (2x the WMMA density
//   per LDS byte vs a 16x32 wave tile).
// ---------------------------------------------------------------------------
#define BM 64
#define BN 128
#define BK 32
#define LDT (BK + 8)

// A-fragment (bf16 16x32): lane l holds row M=l&15,
// K = koff + e, koff = (l>=16)*16.
// B-fragment (bf16 32x16): lane l holds col N=l&15, K = kgrp + e, kgrp = (l>=16)*16.
// C/D: VGPR r, lane l -> M = r + 8*(l>=16), N = l&15.

// ---------------------------------------------------------------------------
// Fused gather + bf16-WMMA gate GEMM + sigmoid*sel_t + scatter-atomicAdd
//   C[k][n] = sigmoid( cat[k][:] . gate_W[n][:] + gb[n] ) * trace[ti[k]][n]
//   ctu[ ui[k]/R ][n] += C[k][n]
// ---------------------------------------------------------------------------
__global__ __launch_bounds__(256) void gate_gemm(
    const float* __restrict__ code, const float* __restrict__ trace,
    const __bf16* __restrict__ gw, const float* __restrict__ gb,
    const int* __restrict__ ci, const int* __restrict__ ti,
    const int* __restrict__ ui, float* __restrict__ ctu)
{
    __shared__ __bf16 As[BM][LDT];
    __shared__ __bf16 Bs[BN][LDT];
    __shared__ int ciS[BM], tiS[BM], orS[BM];

    const int kbase = blockIdx.x * BM;   // gathered-row base
    const int nbase = blockIdx.y * BN;   // gate-output-column base
    const int tid   = threadIdx.x;

    if (tid < BM) {
        int k = kbase + tid;
        ciS[tid] = ci[k];
        tiS[tid] = ti[k];
        orS[tid] = ui[k] >> 3;           // /R (R==8)
    }
    __syncthreads();

    const int wave = tid >> 5;
    const int lane = tid & 31;
    const int wm   = wave & 1;           // M wave-tile (32 rows)
    const int wn   = wave >> 1;          // N wave-tile (32 cols)
    const int mA   = lane & 15;
    const int koffA = (lane >> 4) << 4;
    const int kgrpB = (lane >> 4) << 4;
    const int nB    = lane & 15;

    v8f acc00 = {}, acc01 = {}, acc10 = {}, acc11 = {};

    const int rA  = tid >> 2;            // A staging row 0..63
    const int cA0 = (tid & 3) * 8;
    const int rB  = tid >> 1;            // B staging row 0..127
    const int cB0 = (tid & 1) * 16;

    for (int kb = 0; kb < TWO_D; kb += BK) {
        // ---- stage A (gathered cat rows) and B (gate_W rows), k-major ----
        {
            int crow = ciS[rA], trow = tiS[rA];
            #pragma unroll
            for (int c = 0; c < 8; ++c) {
                int j = kb + cA0 + c;
                float v = (j < DD) ? code[crow * DD + j]
                                   : trace[trow * DD + (j - DD)];
                As[rA][cA0 + c] = (__bf16)v;
            }
            const __bf16* gwp = gw + (nbase + rB) * TWO_D + kb + cB0;
            #pragma unroll
            for (int c = 0; c < 16; ++c) Bs[rB][cB0 + c] = gwp[c];
        }
        __syncthreads();

        // ---- fragments ----
        v16bf a0, a1, b0, b1;
        #pragma unroll
        for (int e = 0; e < 16; ++e) {
            int kkA = koffA + e;
            a0[e] = As[wm * 32 + mA][kkA];
            a1[e] = As[wm * 32 + 16 + mA][kkA];
        }
        #pragma unroll
        for (int e = 0; e < 16; ++e) {
            b0[e] = Bs[wn * 32 + nB][kgrpB + e];
            b1[e] = Bs[wn * 32 + 16 + nB][kgrpB + e];
        }

        acc00 = __builtin_amdgcn_wmma_f32_16x16x32_bf16(false, a0, false, b0,
                                                        (short)0, acc00, false, false);
        acc01 = __builtin_amdgcn_wmma_f32_16x16x32_bf16(false, a0, false, b1,
                                                        (short)0, acc01, false, false);
        acc10 = __builtin_amdgcn_wmma_f32_16x16x32_bf16(false, a1, false, b0,
                                                        (short)0, acc10, false, false);
        acc11 = __builtin_amdgcn_wmma_f32_16x16x32_bf16(false, a1, false, b1,
                                                        (short)0, acc11, false, false);
        __syncthreads();
    }

    // ---- epilogue: sigmoid (rcp-based), scale by sel_t, scatter-accumulate ----
    const int mBaseC = (lane >> 4) * 8;
    const int nC     = lane & 15;
    const int col0   = nbase + wn * 32 + nC;
    const int col1   = col0 + 16;
    const float gb0 = gb[col0], gb1 = gb[col1];

    #pragma unroll
    for (int r = 0; r < 8; ++r) {
        int mloc0 = wm * 32 + mBaseC + r;        // acc0x rows
        int mloc1 = mloc0 + 16;                  // acc1x rows
        {
            int trow = tiS[mloc0], orow = orS[mloc0];
            float g0 = sigmoidf_(acc00[r] + gb0);
            float g1 = sigmoidf_(acc01[r] + gb1);
            atomicAdd(&ctu[orow * DD + col0], g0 * trace[trow * DD + col0]);
            atomicAdd(&ctu[orow * DD + col1], g1 * trace[trow * DD + col1]);
        }
        {
            int trow = tiS[mloc1], orow = orS[mloc1];
            float g0 = sigmoidf_(acc10[r] + gb0);
            float g1 = sigmoidf_(acc11[r] + gb1);
            atomicAdd(&ctu[orow * DD + col0], g0 * trace[trow * DD + col0]);
            atomicAdd(&ctu[orow * DD + col1], g1 * trace[trow * DD + col1]);
        }
    }
}

// ---------------------------------------------------------------------------
// xp = x @ Wih^T + b   (A = x_bf16 [4096x1024], B = Wih_bf16 [1024x1024])
// ---------------------------------------------------------------------------
__global__ __launch_bounds__(256) void xp_gemm(
    const __bf16* __restrict__ xbf, const __bf16* __restrict__ wih,
    const float* __restrict__ bias, float* __restrict__ xp)
{
    __shared__ __bf16 As[BM][LDT];
    __shared__ __bf16 Bs[BN][LDT];

    const int rbase = blockIdx.x * BM;
    const int nbase = blockIdx.y * BN;
    const int tid   = threadIdx.x;

    const int wave = tid >> 5;
    const int lane = tid & 31;
    const int wm   = wave & 1;
    const int wn   = wave >> 1;
    const int mA   = lane & 15;
    const int koffA = (lane >> 4) << 4;
    const int kgrpB = (lane >> 4) << 4;
    const int nB    = lane & 15;

    v8f acc00 = {}, acc01 = {}, acc10 = {}, acc11 = {};

    const int rA  = tid >> 2;
    const int cA0 = (tid & 3) * 8;
    const int rB  = tid >> 1;
    const int cB0 = (tid & 1) * 16;

    for (int kb = 0; kb < TWO_D; kb += BK) {
        const __bf16* ap = xbf + (rbase + rA) * TWO_D + kb + cA0;
        const __bf16* bp = wih + (nbase + rB) * TWO_D + kb + cB0;
        #pragma unroll
        for (int c = 0; c < 8; ++c)  As[rA][cA0 + c] = ap[c];
        #pragma unroll
        for (int c = 0; c < 16; ++c) Bs[rB][cB0 + c] = bp[c];
        __syncthreads();

        v16bf a0, a1, b0, b1;
        #pragma unroll
        for (int e = 0; e < 16; ++e) {
            int kkA = koffA + e;
            a0[e] = As[wm * 32 + mA][kkA];
            a1[e] = As[wm * 32 + 16 + mA][kkA];
        }
        #pragma unroll
        for (int e = 0; e < 16; ++e) {
            b0[e] = Bs[wn * 32 + nB][kgrpB + e];
            b1[e] = Bs[wn * 32 + 16 + nB][kgrpB + e];
        }

        acc00 = __builtin_amdgcn_wmma_f32_16x16x32_bf16(false, a0, false, b0,
                                                        (short)0, acc00, false, false);
        acc01 = __builtin_amdgcn_wmma_f32_16x16x32_bf16(false, a0, false, b1,
                                                        (short)0, acc01, false, false);
        acc10 = __builtin_amdgcn_wmma_f32_16x16x32_bf16(false, a1, false, b0,
                                                        (short)0, acc10, false, false);
        acc11 = __builtin_amdgcn_wmma_f32_16x16x32_bf16(false, a1, false, b1,
                                                        (short)0, acc11, false, false);
        __syncthreads();
    }

    const int mBaseC = (lane >> 4) * 8;
    const int nC     = lane & 15;
    const int col0   = nbase + wn * 32 + nC;
    const int col1   = col0 + 16;
    const float bi0 = bias[col0], bi1 = bias[col1];

    #pragma unroll
    for (int r = 0; r < 8; ++r) {
        int grow0 = rbase + wm * 32 + mBaseC + r;
        int grow1 = grow0 + 16;
        xp[grow0 * FOUR_H + col0] = acc00[r] + bi0;
        xp[grow0 * FOUR_H + col1] = acc01[r] + bi1;
        xp[grow1 * FOUR_H + col0] = acc10[r] + bi0;
        xp[grow1 * FOUR_H + col1] = acc11[r] + bi1;
    }
}

// ---------------------------------------------------------------------------
// Bidirectional LSTM scan. gridDim.x == 2 (block 0 = forward, 1 = backward)
// so the two directions run concurrently. 1024 threads: thread i owns gate
// output i of the 4H matvec; Whh streamed from L2 as packed bf16 (transposed
// layout for coalescing), h/c kept f32 in LDS. Fast rcp-based activations
// keep the sequential critical path short.
// Writes out = code + h directly (fwd -> cols [0,256), bwd -> cols [256,512)).
// ---------------------------------------------------------------------------
__global__ __launch_bounds__(1024) void lstm_scan(
    const float* __restrict__ xpF, const float* __restrict__ xpB,
    const unsigned int* __restrict__ w2F, const unsigned int* __restrict__ w2B,
    const float* __restrict__ code, float* __restrict__ out)
{
    const int dir = blockIdx.x;                  // 0 = fwd, 1 = bwd
    const float* __restrict__ xp = dir ? xpB : xpF;
    const unsigned int* __restrict__ w2 = dir ? w2B : w2F;

    __shared__ float h_s[HH];
    __shared__ float c_s[HH];
    __shared__ float g_s[FOUR_H];

    const int i = threadIdx.x;
    if (i < HH) { h_s[i] = 0.0f; c_s[i] = 0.0f; }
    __syncthreads();

    for (int s = 0; s < NN; ++s) {
        const int t = dir ? (NN - 1 - s) : s;

        // g[i] = xp[t][i] + Whh[i][:] . h   (two partial sums for ILP)
        float acc0 = xp[t * FOUR_H + i];
        float acc1 = 0.0f;
        #pragma unroll 8
        for (int jj = 0; jj < HH / 2; jj += 2) {
            bf16x2 w0 = __builtin_bit_cast(bf16x2, w2[jj * FOUR_H + i]);
            bf16x2 w1 = __builtin_bit_cast(bf16x2, w2[(jj + 1) * FOUR_H + i]);
            acc0 += (float)w0.x * h_s[2 * jj]     + (float)w0.y * h_s[2 * jj + 1];
            acc1 += (float)w1.x * h_s[2 * jj + 2] + (float)w1.y * h_s[2 * jj + 3];
        }
        g_s[i] = acc0 + acc1;
        __syncthreads();

        if (i < HH) {
            float ig = sigmoidf_(g_s[i]);
            float fg = sigmoidf_(g_s[HH + i]);
            float gg = tanhf_(g_s[2 * HH + i]);
            float og = sigmoidf_(g_s[3 * HH + i]);
            float c  = fg * c_s[i] + ig * gg;
            c_s[i] = c;
            float h = og * tanhf_(c);
            h_s[i] = h;
            out[t * DD + dir * HH + i] = code[t * DD + dir * HH + i] + h;
        }
        __syncthreads();
    }
}

// ---------------------------------------------------------------------------
// launcher
// ---------------------------------------------------------------------------
extern "C" void kernel_launch(void* const* d_in, const int* in_sizes, int n_in,
                              void* d_out, int out_size, void* d_ws, size_t ws_size,
                              hipStream_t stream) {
    (void)in_sizes; (void)n_in; (void)out_size; (void)ws_size;

    const float* code   = (const float*)d_in[0];
    const float* trace  = (const float*)d_in[1];
    const float* gate_W = (const float*)d_in[2];
    const float* gate_b = (const float*)d_in[3];
    const float* Wih_f  = (const float*)d_in[4];
    const float* Whh_f  = (const float*)d_in[5];
    const float* b_f    = (const float*)d_in[6];
    const float* Wih_b  = (const float*)d_in[7];
    const float* Whh_b  = (const float*)d_in[8];
    const float* b_b    = (const float*)d_in[9];
    const int*   ci     = (const int*)d_in[10];
    const int*   ti     = (const int*)d_in[11];
    const int*   ui     = (const int*)d_in[12];

    float* out = (float*)d_out;

    // workspace layout (bytes)
    char* ws = (char*)d_ws;
    float*        ctu  = (float*)(ws + 0);                     //  8 MB: [N][D]
    __bf16*       xbf  = (__bf16*)(ws + (8u << 20));           //  8 MB: [N][2D]
    __bf16*       gwbf = (__bf16*)(ws + (16u << 20));          //  1 MB
    __bf16*       wihF = (__bf16*)(ws + (17u << 20));          //  2 MB
    __bf16*       wihB = (__bf16*)(ws + (19u << 20));          //  2 MB
    unsigned int* w2F  = (unsigned int*)(ws + (21u << 20));    // .5 MB
    unsigned int* w2B  = (unsigned int*)(ws + (21u << 20) + (512u << 10));
    float*        xpF  = (float*)(ws + (22u << 20));           // 16 MB: [N][4H]
    float*        xpB  = (float*)(ws + (38u << 20));           // 16 MB

    // 1) zero the scatter-reduce accumulator
    zero_f32<<<(NN * DD + 255) / 256, 256, 0, stream>>>(ctu, NN * DD);

    // 2) bf16 weight conversions (L2-resident afterwards)
    f32_to_bf16<<<(DD * TWO_D + 255) / 256, 256, 0, stream>>>(gate_W, gwbf, DD * TWO_D);
    f32_to_bf16<<<(FOUR_H * TWO_D + 255) / 256, 256, 0, stream>>>(Wih_f, wihF, FOUR_H * TWO_D);
    f32_to_bf16<<<(FOUR_H * TWO_D + 255) / 256, 256, 0, stream>>>(Wih_b, wihB, FOUR_H * TWO_D);
    pack_whh<<<(128 * FOUR_H + 255) / 256, 256, 0, stream>>>(Whh_f, w2F);
    pack_whh<<<(128 * FOUR_H + 255) / 256, 256, 0, stream>>>(Whh_b, w2B);

    // 3) fused gather + WMMA gate GEMM + sigmoid*sel_t + scatter-atomicAdd
    gate_gemm<<<dim3(KK / BM, DD / BN), 256, 0, stream>>>(
        code, trace, gwbf, gate_b, ci, ti, ui, ctu);

    // 4) x = [code | ctu] in bf16
    build_x<<<(NN * TWO_D + 255) / 256, 256, 0, stream>>>(code, ctu, xbf);

    // 5) WMMA input projections for both directions
    xp_gemm<<<dim3(NN / BM, FOUR_H / BN), 256, 0, stream>>>(xbf, wihF, b_f, xpF);
    xp_gemm<<<dim3(NN / BM, FOUR_H / BN), 256, 0, stream>>>(xbf, wihB, b_b, xpB);

    // 6) bidirectional sequential scan (both directions concurrently)
    lstm_scan<<<2, 1024, 0, stream>>>(xpF, xpB, w2F, w2B, code, out);
}
